// TransformerModel_25589415150279
// MI455X (gfx1250) — compile-verified, matrix operand order checked by reading, not run on hardware
//
#include <hip/hip_runtime.h>
#include <hip/hip_bf16.h>

// ---------------------------------------------------------------------------
// CDNA5 (gfx1250, wave32) transformer forward pass.
// Core engine: batched tiled GEMM using v_wmma_f32_16x16x32_f16.
//   - block = 256 threads = 8 waves; block tile 128x64; wave tile 32x32
//   - K stepped by 32; f32 global operands converted to f16 in LDS staging
//   - double-buffered LDS + register prefetch; last K-step peeled so the
//     steady-state loop is branch-free: load-next / compute / store-next /
//     one s_barrier.  TRANSB is a template parameter.
//   - __builtin_amdgcn_sched_barrier(0) fences keep the prefetch
//     global_loads issued BEFORE the WMMA cluster and the cvt/ds_store
//     (with its loadcnt wait) AFTER it, so loads overlap XDL execution.
//   - fragment layout per CDNA5 ISA (16-bit A 16x32 / B 32x16):
//       lane&15 selects row(A)/col(B), lane>>4 selects K-octet,
//       v16h elems 0..7 -> K = half*8+0..7, elems 8..15 -> K = 16+half*8+0..7
//     => two aligned 16B LDS loads per fragment.
// ---------------------------------------------------------------------------

typedef _Float16 v8h  __attribute__((ext_vector_type(8)));
typedef _Float16 v16h __attribute__((ext_vector_type(16)));
typedef float    v8f  __attribute__((ext_vector_type(8)));

#define LDSPAD 40   // 32 halves + 8 pad (row stride 80B, keeps 16B alignment)

template <bool TRANSB>
__global__ __launch_bounds__(256) void gemm_wmma_f16(
    const float* __restrict__ A, const float* __restrict__ B,
    float* __restrict__ C, const float* __restrict__ bias,
    int M, int N, int K, int lda, int ldb, int ldc,
    long sAb, long sAh, long sBb, long sBh, long sCb, long sCh,
    int Hdim, int relu)
{
  __shared__ _Float16 As[2][128 * LDSPAD];
  __shared__ _Float16 Bs[2][64 * LDSPAD];

  const int z  = blockIdx.z;
  const int bb = z / Hdim;
  const int hh = z - bb * Hdim;
  A += (size_t)bb * sAb + (size_t)hh * sAh;
  B += (size_t)bb * sBb + (size_t)hh * sBh;
  C += (size_t)bb * sCb + (size_t)hh * sCh;

  const int n0 = blockIdx.x * 64;
  const int m0 = blockIdx.y * 128;
  const int tid    = threadIdx.x;
  const int lane   = tid & 31;
  const int wave   = tid >> 5;
  const int wm     = (wave >> 1) * 32;   // wave tile origin (rows), 0..96
  const int wn     = (wave & 1) * 32;    // wave tile origin (cols), 0/32
  const int rr     = lane & 15;
  const int halfId = lane >> 4;

  // ---- staging helpers (register prefetch + LDS store) ----
  // A tile: 128 rows x 32 k = 1024 float4 slots; thread covers tid + i*256.
  auto loadA = [&](int k0, float4* pa) {
    #pragma unroll
    for (int i = 0; i < 4; ++i) {
      int s = tid + i * 256;
      int r = s >> 3, c = (s & 7) << 2;
      pa[i] = *(const float4*)(A + (size_t)(m0 + r) * lda + (k0 + c));
    }
  };
  auto storeA = [&](int buf, const float4* pa) {
    #pragma unroll
    for (int i = 0; i < 4; ++i) {
      int s = tid + i * 256;
      int r = s >> 3, c = (s & 7) << 2;
      _Float16* d = &As[buf][r * LDSPAD + c];
      d[0] = (_Float16)pa[i].x; d[1] = (_Float16)pa[i].y;
      d[2] = (_Float16)pa[i].z; d[3] = (_Float16)pa[i].w;
    }
  };
  // B tile staged as [col n][k].
  //   TRANSB: B is [N,K] row-major -> direct copy (512 float4 slots, 2/thread)
  //   else  : B is [K,N] row-major -> transpose while staging (8 scalars/thread)
  auto loadB = [&](int k0, float* pb) {
    if (TRANSB) {
      #pragma unroll
      for (int i = 0; i < 2; ++i) {
        int s = tid + i * 256;
        int r = s >> 3, c = (s & 7) << 2;
        float4 v = *(const float4*)(B + (size_t)(n0 + r) * ldb + (k0 + c));
        pb[i * 4 + 0] = v.x; pb[i * 4 + 1] = v.y;
        pb[i * 4 + 2] = v.z; pb[i * 4 + 3] = v.w;
      }
    } else {
      int n = tid & 63, kb = (tid >> 6) << 3;
      #pragma unroll
      for (int i = 0; i < 8; ++i)
        pb[i] = B[(size_t)(k0 + kb + i) * ldb + (n0 + n)];
    }
  };
  auto storeB = [&](int buf, const float* pb) {
    if (TRANSB) {
      #pragma unroll
      for (int i = 0; i < 2; ++i) {
        int s = tid + i * 256;
        int r = s >> 3, c = (s & 7) << 2;
        _Float16* d = &Bs[buf][r * LDSPAD + c];
        d[0] = (_Float16)pb[i * 4 + 0]; d[1] = (_Float16)pb[i * 4 + 1];
        d[2] = (_Float16)pb[i * 4 + 2]; d[3] = (_Float16)pb[i * 4 + 3];
      }
    } else {
      int n = tid & 63, kb = (tid >> 6) << 3;
      #pragma unroll
      for (int i = 0; i < 8; ++i)
        Bs[buf][n * LDSPAD + kb + i] = (_Float16)pb[i];
    }
  };

  v8f acc[2][2];
  #pragma unroll
  for (int mi = 0; mi < 2; ++mi)
    #pragma unroll
    for (int ni = 0; ni < 2; ++ni)
      #pragma unroll
      for (int e = 0; e < 8; ++e) acc[mi][ni][e] = 0.0f;

  auto computeTile = [&](int cur) {
    v16h af[2], bf[2];
    #pragma unroll
    for (int mi = 0; mi < 2; ++mi) {
      const _Float16* p = &As[cur][(wm + mi * 16 + rr) * LDSPAD + halfId * 8];
      v8h lo = *(const v8h*)p;
      v8h hi = *(const v8h*)(p + 16);
      af[mi] = __builtin_shufflevector(lo, hi, 0,1,2,3,4,5,6,7,8,9,10,11,12,13,14,15);
    }
    #pragma unroll
    for (int ni = 0; ni < 2; ++ni) {
      const _Float16* p = &Bs[cur][(wn + ni * 16 + rr) * LDSPAD + halfId * 8];
      v8h lo = *(const v8h*)p;
      v8h hi = *(const v8h*)(p + 16);
      bf[ni] = __builtin_shufflevector(lo, hi, 0,1,2,3,4,5,6,7,8,9,10,11,12,13,14,15);
    }
    #pragma unroll
    for (int mi = 0; mi < 2; ++mi)
      #pragma unroll
      for (int ni = 0; ni < 2; ++ni)
        acc[mi][ni] = __builtin_amdgcn_wmma_f32_16x16x32_f16(
            false, af[mi], false, bf[ni], (short)0, acc[mi][ni], false, false);
  };

  float4 pa[4];
  float  pb[8];

  // prologue: stage first tile
  loadA(0, pa); loadB(0, pb);
  storeA(0, pa); storeB(0, pb);
  __syncthreads();

  // steady state: branch-free load-next / compute / store-next / barrier.
  // sched_barrier(0) fences pin: loads issue before the WMMA cluster; the
  // cvt+ds_store chain (carrying the loadcnt wait) stays after it.
  int cur = 0;
  for (int k0 = 32; k0 < K; k0 += 32) {
    loadA(k0, pa); loadB(k0, pb);        // next tile's global loads in flight
    __builtin_amdgcn_sched_barrier(0);
    computeTile(cur);                    // 4 WMMAs off current LDS buffer
    __builtin_amdgcn_sched_barrier(0);
    storeA(cur ^ 1, pa);                 // loadcnt wait overlaps XDL above
    storeB(cur ^ 1, pb);
    __syncthreads();
    cur ^= 1;
  }
  computeTile(cur);                      // peeled last K-step

  // ---- epilogue: bias / relu / store f32 ----
  #pragma unroll
  for (int mi = 0; mi < 2; ++mi) {
    #pragma unroll
    for (int ni = 0; ni < 2; ++ni) {
      int n = n0 + wn + ni * 16 + rr;
      float bv = bias ? bias[n] : 0.0f;
      #pragma unroll
      for (int e = 0; e < 8; ++e) {
        int m = m0 + wm + mi * 16 + halfId * 8 + e;
        float v = acc[mi][ni][e] + bv;
        if (relu) v = fmaxf(v, 0.0f);
        C[(size_t)m * ldc + n] = v;
      }
    }
  }
}

// ---- masked softmax over score rows (S = 512, 2 elems/thread) --------------
__global__ __launch_bounds__(256) void softmax_mask_kernel(
    float* __restrict__ scores, const int* __restrict__ src,
    int S, int Hdim, int causal, float scale)
{
  const int q = blockIdx.x;
  const int z = blockIdx.y;
  const int b = z / Hdim;
  float* row = scores + ((size_t)z * S + q) * S;
  const int tid = threadIdx.x;
  __shared__ float red[256];

  float vals[2], mx = -3.0e38f;
  #pragma unroll
  for (int j = 0; j < 2; ++j) {
    int k = tid + j * 256;
    float v = row[k] * scale;
    bool valid = causal ? (k <= q) : (src[b * S + k] != 0);
    v = valid ? v : -1.0e9f;
    vals[j] = v;
    mx = fmaxf(mx, v);
  }
  red[tid] = mx; __syncthreads();
  for (int o = 128; o > 0; o >>= 1) { if (tid < o) red[tid] = fmaxf(red[tid], red[tid + o]); __syncthreads(); }
  mx = red[0]; __syncthreads();

  float s = 0.0f;
  #pragma unroll
  for (int j = 0; j < 2; ++j) { vals[j] = __expf(vals[j] - mx); s += vals[j]; }
  red[tid] = s; __syncthreads();
  for (int o = 128; o > 0; o >>= 1) { if (tid < o) red[tid] += red[tid + o]; __syncthreads(); }
  float inv = 1.0f / red[0];
  #pragma unroll
  for (int j = 0; j < 2; ++j) row[tid + j * 256] = vals[j] * inv;
}

// ---- y = LayerNorm(x + t) * g + b, in place on x (D = 1024) ----------------
__global__ __launch_bounds__(256) void add_ln_kernel(
    float* __restrict__ x, const float* __restrict__ t,
    const float* __restrict__ g, const float* __restrict__ b, int D)
{
  float* xr = x + (size_t)blockIdx.x * D;
  const float* tr = t + (size_t)blockIdx.x * D;
  const int tid = threadIdx.x;
  __shared__ float red[256];

  float v[4], s = 0.0f;
  #pragma unroll
  for (int j = 0; j < 4; ++j) { int d = tid + j * 256; v[j] = xr[d] + tr[d]; s += v[j]; }
  red[tid] = s; __syncthreads();
  for (int o = 128; o > 0; o >>= 1) { if (tid < o) red[tid] += red[tid + o]; __syncthreads(); }
  float mean = red[0] / (float)D; __syncthreads();

  float s2 = 0.0f;
  #pragma unroll
  for (int j = 0; j < 4; ++j) { float dm = v[j] - mean; s2 += dm * dm; }
  red[tid] = s2; __syncthreads();
  for (int o = 128; o > 0; o >>= 1) { if (tid < o) red[tid] += red[tid + o]; __syncthreads(); }
  float rstd = rsqrtf(red[0] / (float)D + 1e-5f);

  #pragma unroll
  for (int j = 0; j < 4; ++j) {
    int d = tid + j * 256;
    xr[d] = (v[j] - mean) * rstd * g[d] + b[d];
  }
}

// ---- out[row] = emb[tok[row]] + sin/cos positional encoding ----------------
__global__ __launch_bounds__(256) void embed_kernel(
    float* __restrict__ out, const int* __restrict__ tok,
    const float* __restrict__ emb, int S, int D)
{
  const int row = blockIdx.x;
  const int s   = row % S;
  const int t   = tok[row];
  const int tid = threadIdx.x;
  const float c = -9.2103403719761836f / (float)D;   // -ln(10000)/D
  #pragma unroll
  for (int j = 0; j < 4; ++j) {
    int d  = tid + j * 256;
    int de = d & ~1;
    float div = __expf((float)de * c);
    float ang = (float)s * div;
    float pe  = (d & 1) ? __cosf(ang) : __sinf(ang);
    out[(size_t)row * D + d] = emb[(size_t)t * D + d] + pe;
  }
}

// ---- pooled[b] = mean over S of x[b,:,:] -----------------------------------
__global__ __launch_bounds__(256) void meanpool_kernel(
    const float* __restrict__ x, float* __restrict__ pooled, int S, int D)
{
  const int b = blockIdx.x, tid = threadIdx.x;
  #pragma unroll
  for (int j = 0; j < 4; ++j) {
    int d = tid + j * 256;
    float s = 0.0f;
    for (int ss = 0; ss < S; ++ss) s += x[((size_t)b * S + ss) * D + d];
    pooled[(size_t)b * D + d] = s / (float)S;
  }
}

// ---- out[b] = pooled[b] @ fcW + fcb  (D=1024, C=4) -------------------------
__global__ __launch_bounds__(256) void fc_kernel(
    const float* __restrict__ pooled, const float* __restrict__ W,
    const float* __restrict__ bias, float* __restrict__ out, int D, int C)
{
  const int b = blockIdx.x, tid = threadIdx.x;
  __shared__ float red[256];
  for (int c = 0; c < C; ++c) {
    float s = 0.0f;
    for (int d = tid; d < D; d += 256) s += pooled[(size_t)b * D + d] * W[(size_t)d * C + c];
    red[tid] = s; __syncthreads();
    for (int o = 128; o > 0; o >>= 1) { if (tid < o) red[tid] += red[tid + o]; __syncthreads(); }
    if (tid == 0) out[b * C + c] = red[0] + bias[c];
    __syncthreads();
  }
}

// ---------------------------------------------------------------------------
extern "C" void kernel_launch(void* const* d_in, const int* in_sizes, int n_in,
                              void* d_out, int out_size, void* d_ws, size_t ws_size,
                              hipStream_t stream)
{
  (void)in_sizes; (void)out_size; (void)ws_size;
  const int Bb = 16, Ss = 512, Dd = 1024, Hh = 16, Ff = 4096, Cc = 4;
  const int ROWS = Bb * Ss;                    // 8192
  if (n_in < 257) return;

  const int*   src = (const int*)d_in[0];
  const int*   tgt = (const int*)d_in[1];
  const float* emb = (const float*)d_in[2];
  auto P = [&](int i) { return (const float*)d_in[i]; };
  // setup_inputs() insertion order:
  //   enc layer l base = 3 + l*16 : Wq bq Wk bk Wv bv Wo bo W1 b1 W2 b2 g1 b1n g2 b2n
  //   dec layer l base = 99 + l*26: self[8] cross[8] W1 b1 W2 b2 g1 b1n g2 b2n g3 b3n
  //   fcW = 255, fcb = 256

  // workspace carve-up (f32)
  char* wp = (char*)d_ws;
  auto carve = [&](size_t bytes) { float* p = (float*)wp; wp += (bytes + 255) & ~(size_t)255; return p; };
  float* Xe = carve((size_t)ROWS * Dd * 4);            // encoder activations (memory)
  float* Xd = carve((size_t)ROWS * Dd * 4);            // decoder activations
  float* Qb = carve((size_t)ROWS * Dd * 4);            // Q / generic result
  float* Kb = carve((size_t)ROWS * Dd * 4);            // K
  float* Vb = carve((size_t)ROWS * Dd * 4);            // V
  float* Tb = carve((size_t)ROWS * Dd * 4);            // attn context
  float* Hb = carve((size_t)ROWS * Ff * 4);            // FFN hidden
  float* Sc = carve((size_t)Bb * Hh * Ss * Ss * 4);    // attention scores
  float* Pl = carve((size_t)Bb * Dd * 4);              // pooled

  auto gemm = [&](const float* A, const float* B, float* C, const float* bias,
                  int M, int N, int K, int lda, int ldb, int ldc,
                  long sAb, long sAh, long sBb, long sBh, long sCb, long sCh,
                  int Hdim, int batch, int transB, int relu) {
    dim3 grid(N / 64, M / 128, batch);
    if (transB)
      gemm_wmma_f16<true><<<grid, dim3(256), 0, stream>>>(
          A, B, C, bias, M, N, K, lda, ldb, ldc,
          sAb, sAh, sBb, sBh, sCb, sCh, Hdim, relu);
    else
      gemm_wmma_f16<false><<<grid, dim3(256), 0, stream>>>(
          A, B, C, bias, M, N, K, lda, ldb, ldc,
          sAb, sAh, sBb, sBh, sCb, sCh, Hdim, relu);
  };

  const long SD = (long)Ss * Dd, SS = (long)Ss * Ss;

  // multi-head attention: result (pre-residual) left in Qb
  auto attention = [&](const float* xq, const float* xkv, int pb, int causal) {
    gemm(xq,  P(pb + 0), Qb, P(pb + 1), ROWS, Dd, Dd, Dd, Dd, Dd, 0,0,0,0,0,0, 1, 1, 0, 0);
    gemm(xkv, P(pb + 2), Kb, P(pb + 3), ROWS, Dd, Dd, Dd, Dd, Dd, 0,0,0,0,0,0, 1, 1, 0, 0);
    gemm(xkv, P(pb + 4), Vb, P(pb + 5), ROWS, Dd, Dd, Dd, Dd, Dd, 0,0,0,0,0,0, 1, 1, 0, 0);
    // scores[b,h] = Q_bh (S x 64) @ K_bh^T (64 x S)
    gemm(Qb, Kb, Sc, nullptr, Ss, Ss, 64, Dd, Dd, Ss,
         SD, 64, SD, 64, (long)Hh * SS, SS, Hh, Bb * Hh, 1, 0);
    softmax_mask_kernel<<<dim3(Ss, Bb * Hh), 256, 0, stream>>>(Sc, src, Ss, Hh, causal, 0.125f);
    // ctx[b,h] = P (S x S) @ V_bh (S x 64); C base b*S*D + h*64 -> [b,s,h*dk+d]
    gemm(Sc, Vb, Tb, nullptr, Ss, 64, Ss, Ss, Dd, Dd,
         (long)Hh * SS, SS, SD, 64, SD, 64, Hh, Bb * Hh, 0, 0);
    gemm(Tb, P(pb + 6), Qb, P(pb + 7), ROWS, Dd, Dd, Dd, Dd, Dd, 0,0,0,0,0,0, 1, 1, 0, 0);
  };

  auto ffn = [&](float* X, int pb) {  // pb -> W1; result in Qb
    gemm(X,  P(pb + 0), Hb, P(pb + 1), ROWS, Ff, Dd, Dd, Ff, Ff, 0,0,0,0,0,0, 1, 1, 0, 1);
    gemm(Hb, P(pb + 2), Qb, P(pb + 3), ROWS, Dd, Ff, Ff, Dd, Dd, 0,0,0,0,0,0, 1, 1, 0, 0);
  };
  auto add_ln = [&](float* X, int gi, int bi) {
    add_ln_kernel<<<dim3(ROWS), 256, 0, stream>>>(X, Qb, P(gi), P(bi), Dd);
  };

  // embeddings + positional encoding
  embed_kernel<<<dim3(ROWS), 256, 0, stream>>>(Xe, src, emb, Ss, Dd);
  embed_kernel<<<dim3(ROWS), 256, 0, stream>>>(Xd, tgt, emb, Ss, Dd);

  // encoder
  for (int l = 0; l < 6; ++l) {
    int base = 3 + l * 16;
    attention(Xe, Xe, base, /*causal=*/0);
    add_ln(Xe, base + 12, base + 13);
    ffn(Xe, base + 8);
    add_ln(Xe, base + 14, base + 15);
  }
  // decoder
  for (int l = 0; l < 6; ++l) {
    int base = 99 + l * 26;
    attention(Xd, Xd, base + 0, /*causal=*/1);        // self
    add_ln(Xd, base + 20, base + 21);
    attention(Xd, Xe, base + 8, /*causal=*/0);        // cross (src padding mask)
    add_ln(Xd, base + 22, base + 23);
    ffn(Xd, base + 16);
    add_ln(Xd, base + 24, base + 25);
  }

  meanpool_kernel<<<dim3(Bb), 256, 0, stream>>>(Xd, Pl, Ss, Dd);
  fc_kernel<<<dim3(Bb), 256, 0, stream>>>(Pl, P(255), P(256), (float*)d_out, Dd, Cc);
}
